// DSTransformerIP_38010460569643
// MI455X (gfx1250) — compile-verified, matrix operand order checked by reading, not run on hardware
//
#include <hip/hip_runtime.h>
#include <hip/hip_bf16.h>
#include <math.h>

// ---------------------------------------------------------------------------
// Model constants
// ---------------------------------------------------------------------------
#define BB    8
#define NTOK  4096
#define DIMC  512
#define HH    8
#define HDIM  64
#define MLPC  2048

// ---------------------------------------------------------------------------
// WMMA types
// ---------------------------------------------------------------------------
typedef __attribute__((ext_vector_type(16))) __bf16 v16bf;
typedef __attribute__((ext_vector_type(8)))  float  v8f;
typedef __attribute__((ext_vector_type(2)))  float  v2f;
typedef __attribute__((ext_vector_type(2)))  __bf16 v2bf;

union FragBF {
    v16bf    v;
    unsigned u32[8];
};

// single packed f32x2 -> bf16x2 conversion (lowers to v_cvt_pk_bf16_f32)
__device__ __forceinline__ unsigned pack_bf16x2(float lo, float hi) {
    union { v2bf v; unsigned u; } p;
    v2f t;
    t[0] = lo;
    t[1] = hi;
    p.v = __builtin_convertvector(t, v2bf);
    return p.u;
}

// ---------------------------------------------------------------------------
// WMMA GEMM:  C = act(A @ W + bias) [+ residual]
//   A: (M,K) f32 row-major, W: (K,N) f32 row-major, C: (M,N) f32
//   M % 64 == 0, K % 32 == 0, N % 256 == 0  (guaranteed by the model dims)
// Block tile 64x256, 256 threads = 8 waves (2x4); each wave owns 32x64
// (2x4 accumulators -> 8 v_wmma per K-chunk from 2 A + 4 B fragments).
// Register-prefetch pipeline overlaps next chunk's global loads with WMMAs.
// All LDS staging stores are packed 2xbf16 dwords (native v_cvt_pk_bf16_f32).
// ---------------------------------------------------------------------------
#define GT_M 64
#define GT_N 256
#define GT_K 32
#define LDSS 36   // padded bf16 stride (72B) -> conflict-free fragment reads

template <int ACT, bool HAS_BIAS, bool HAS_RES>
__global__ __launch_bounds__(256)
void gemm_wmma_kernel(const float* __restrict__ A, const float* __restrict__ W,
                      const float* __restrict__ bias, const float* __restrict__ residual,
                      float* __restrict__ C, int M, int K, int Nn)
{
    __shared__ unsigned short As[GT_M][LDSS];   // [m][k]
    __shared__ unsigned short Bs[GT_N][LDSS];   // [n][k] (transposed tile of W)

    const int tid  = threadIdx.x;
    const int lane = tid & 31;
    const int wave = tid >> 5;     // 0..7
    const int wm   = wave >> 2;    // 0..1  (32 rows each)
    const int wn   = wave & 3;     // 0..3  (64 cols each)
    const int bm   = blockIdx.y * GT_M;
    const int bn   = blockIdx.x * GT_N;
    const int half = lane >> 4;
    const int l15  = lane & 15;

    // staging coordinates: A tile 64x32, one row-octet per thread
    const int rA = tid >> 2;           // 0..63  row of A tile
    const int kA = (tid & 3) * 8;      // 0,8,16,24
    // B tile 32x256: per thread 4 iterations of (2 k-adjacent float4 of W)
    //   w = tid + i*256 (0..1023): kB = (w>>6)*2, nB = (w&63)*4

    v8f acc[2][4];
#pragma unroll
    for (int i = 0; i < 2; ++i)
#pragma unroll
        for (int j = 0; j < 4; ++j)
#pragma unroll
            for (int r = 0; r < 8; ++r) acc[i][j][r] = 0.0f;

    const int nch = K / GT_K;

    // ---- prefetch chunk 0 into registers (float4 -> global_load_b128) ----
    float4 aR0, aR1;
    float4 bRa[4], bRb[4];
    {
        const float* as = A + (size_t)(bm + rA) * K + kA;
        aR0 = *(const float4*)(as);
        aR1 = *(const float4*)(as + 4);
#pragma unroll
        for (int i = 0; i < 4; ++i) {
            int w  = tid + i * 256;            // 0..1023
            int kk = (w >> 6) * 2;             // 0,2,..,30
            int nn = (w & 63) * 4;             // 0..252
            bRa[i] = *(const float4*)(W + (size_t)kk * Nn + bn + nn);
            bRb[i] = *(const float4*)(W + (size_t)(kk + 1) * Nn + bn + nn);
        }
    }

    for (int ch = 0; ch < nch; ++ch) {
        // ---- store staged registers to LDS as packed bf16 pairs ----
        *(unsigned*)&As[rA][kA + 0] = pack_bf16x2(aR0.x, aR0.y);
        *(unsigned*)&As[rA][kA + 2] = pack_bf16x2(aR0.z, aR0.w);
        *(unsigned*)&As[rA][kA + 4] = pack_bf16x2(aR1.x, aR1.y);
        *(unsigned*)&As[rA][kA + 6] = pack_bf16x2(aR1.z, aR1.w);
#pragma unroll
        for (int i = 0; i < 4; ++i) {
            int w  = tid + i * 256;
            int kk = (w >> 6) * 2;
            int nn = (w & 63) * 4;
            *(unsigned*)&Bs[nn + 0][kk] = pack_bf16x2(bRa[i].x, bRb[i].x);
            *(unsigned*)&Bs[nn + 1][kk] = pack_bf16x2(bRa[i].y, bRb[i].y);
            *(unsigned*)&Bs[nn + 2][kk] = pack_bf16x2(bRa[i].z, bRb[i].z);
            *(unsigned*)&Bs[nn + 3][kk] = pack_bf16x2(bRa[i].w, bRb[i].w);
        }
        __syncthreads();

        // ---- prefetch next chunk while WMMAs run ----
        if (ch + 1 < nch) {
            const int k0 = (ch + 1) * GT_K;
            const float* as = A + (size_t)(bm + rA) * K + k0 + kA;
            aR0 = *(const float4*)(as);
            aR1 = *(const float4*)(as + 4);
#pragma unroll
            for (int i = 0; i < 4; ++i) {
                int w  = tid + i * 256;
                int kk = (w >> 6) * 2;
                int nn = (w & 63) * 4;
                bRa[i] = *(const float4*)(W + (size_t)(k0 + kk) * Nn + bn + nn);
                bRb[i] = *(const float4*)(W + (size_t)(k0 + kk + 1) * Nn + bn + nn);
            }
        }

        // ---- build fragments (ISA 16-bit A 16x32 layout) ----
        FragBF fa[2], fb[4];
#pragma unroll
        for (int s = 0; s < 2; ++s) {
            const int arow = wm * 32 + s * 16 + l15;
#pragma unroll
            for (int j = 0; j < 8; ++j) {
                const int kp = ((j < 4) ? (2 * j) : (2 * j + 8)) + half * 8;
                fa[s].u32[j] = *(const unsigned*)&As[arow][kp];
            }
        }
#pragma unroll
        for (int t = 0; t < 4; ++t) {
            const int bcol = wn * 64 + t * 16 + l15;
#pragma unroll
            for (int j = 0; j < 8; ++j) {
                const int kp = ((j < 4) ? (2 * j) : (2 * j + 8)) + half * 8;
                fb[t].u32[j] = *(const unsigned*)&Bs[bcol][kp];
            }
        }
#pragma unroll
        for (int s = 0; s < 2; ++s)
#pragma unroll
            for (int t = 0; t < 4; ++t)
                acc[s][t] = __builtin_amdgcn_wmma_f32_16x16x32_bf16(
                    false, fa[s].v, false, fb[t].v, (short)0, acc[s][t], false, false);
        __syncthreads();
    }

    // ---- epilogue: bias / relu / residual, store f32 (branch-free) ----
#pragma unroll
    for (int s = 0; s < 2; ++s) {
#pragma unroll
        for (int t = 0; t < 4; ++t) {
            const int ncol = bn + wn * 64 + t * 16 + l15;
            const float bv = HAS_BIAS ? bias[ncol] : 0.0f;
#pragma unroll
            for (int r = 0; r < 8; ++r) {
                const int mg = bm + wm * 32 + s * 16 + r + half * 8;
                float val = acc[s][t][r] + bv;
                if (ACT == 1) val = fmaxf(val, 0.0f);
                if (HAS_RES) val += residual[(size_t)mg * Nn + ncol];
                C[(size_t)mg * Nn + ncol] = val;
            }
        }
    }
}

// ---------------------------------------------------------------------------
// LayerNorm: one block per row of C=512
// ---------------------------------------------------------------------------
__global__ __launch_bounds__(256)
void ln_kernel(const float* __restrict__ x, const float* __restrict__ g,
               const float* __restrict__ b, float* __restrict__ y, int C)
{
    __shared__ float sh[256];
    const int row = blockIdx.x;
    const int tid = threadIdx.x;
    const float* xr = x + (size_t)row * C;
    float* yr = y + (size_t)row * C;

    float s = 0.0f;
    for (int c = tid; c < C; c += 256) s += xr[c];
    sh[tid] = s; __syncthreads();
    for (int st = 128; st > 0; st >>= 1) { if (tid < st) sh[tid] += sh[tid + st]; __syncthreads(); }
    const float mean = sh[0] / (float)C;
    __syncthreads();

    float v = 0.0f;
    for (int c = tid; c < C; c += 256) { float d = xr[c] - mean; v += d * d; }
    sh[tid] = v; __syncthreads();
    for (int st = 128; st > 0; st >>= 1) { if (tid < st) sh[tid] += sh[tid + st]; __syncthreads(); }
    const float rstd = rsqrtf(sh[0] / (float)C + 1e-5f);

    for (int c = tid; c < C; c += 256)
        yr[c] = (xr[c] - mean) * rstd * g[c] + b[c];
}

// ---------------------------------------------------------------------------
// score: sigmoid(x @ w + b0), one block per row
// ---------------------------------------------------------------------------
__global__ __launch_bounds__(256)
void score_kernel(const float* __restrict__ x, const float* __restrict__ w,
                  const float* __restrict__ b, float* __restrict__ out, int C)
{
    __shared__ float sh[256];
    const int row = blockIdx.x;
    const int tid = threadIdx.x;
    const float* xr = x + (size_t)row * C;
    float s = 0.0f;
    for (int c = tid; c < C; c += 256) s += xr[c] * w[c];
    sh[tid] = s; __syncthreads();
    for (int st = 128; st > 0; st >>= 1) { if (tid < st) sh[tid] += sh[tid + st]; __syncthreads(); }
    if (tid == 0) out[row] = 1.0f / (1.0f + expf(-(sh[0] + b[0])));
}

// ---------------------------------------------------------------------------
// per-row sum of squares
// ---------------------------------------------------------------------------
__global__ __launch_bounds__(256)
void sqnorm_kernel(const float* __restrict__ x, float* __restrict__ out, int C)
{
    __shared__ float sh[256];
    const int row = blockIdx.x;
    const int tid = threadIdx.x;
    const float* xr = x + (size_t)row * C;
    float s = 0.0f;
    for (int c = tid; c < C; c += 256) s += xr[c] * xr[c];
    sh[tid] = s; __syncthreads();
    for (int st = 128; st > 0; st >>= 1) { if (tid < st) sh[tid] += sh[tid + st]; __syncthreads(); }
    if (tid == 0) out[row] = sh[0];
}

// ---------------------------------------------------------------------------
// attention: dots[b,h,i,j] = scale * sum_d q[b,i,h*64+d] * k[b,j,h*64+d]
// ---------------------------------------------------------------------------
__global__ __launch_bounds__(256)
void dots_kernel(const float* __restrict__ q, int ldq,
                 const float* __restrict__ k, int ldk,
                 float* __restrict__ dots, int NQ, int NK, float scale, int BHcnt)
{
    size_t id = (size_t)blockIdx.x * blockDim.x + threadIdx.x;
    size_t total = (size_t)BHcnt * NQ * NK;
    if (id >= total) return;
    int j = (int)(id % NK); size_t r = id / NK;
    int i = (int)(r % NQ);  r /= NQ;
    int h = (int)(r % HH);
    int bb = (int)(r / HH);
    const float* qr = q + ((size_t)bb * NQ + i) * ldq + h * HDIM;
    const float* kr = k + ((size_t)bb * NK + j) * ldk + h * HDIM;
    float s = 0.0f;
#pragma unroll
    for (int d = 0; d < HDIM; ++d) s += qr[d] * kr[d];
    dots[id] = s * scale;
}

// softmax over last dim (plusone: off-by-one softmax denominator)
__global__ __launch_bounds__(256)
void softmax_kernel(float* __restrict__ a, int NK, int plusone)
{
    __shared__ float sh[256];
    const size_t row = blockIdx.x;
    const int tid = threadIdx.x;
    float* ar = a + row * (size_t)NK;
    float m = -3.0e38f;
    for (int j = tid; j < NK; j += 256) m = fmaxf(m, ar[j]);
    sh[tid] = m; __syncthreads();
    for (int st = 128; st > 0; st >>= 1) { if (tid < st) sh[tid] = fmaxf(sh[tid], sh[tid + st]); __syncthreads(); }
    const float mx = sh[0];
    __syncthreads();
    float s = 0.0f;
    for (int j = tid; j < NK; j += 256) { float e = expf(ar[j] - mx); ar[j] = e; s += e; }
    sh[tid] = s; __syncthreads();
    for (int st = 128; st > 0; st >>= 1) { if (tid < st) sh[tid] += sh[tid + st]; __syncthreads(); }
    const float inv = 1.0f / (sh[0] + (plusone ? 1.0f : 0.0f));
    for (int j = tid; j < NK; j += 256) ar[j] *= inv;
}

// out[b,i,h*64+d] = sum_j attn[b,h,i,j] * v[b,j,h*64+d]  (out ld = 512)
__global__ __launch_bounds__(256)
void attnout_kernel(const float* __restrict__ attn, const float* __restrict__ v, int ldv,
                    float* __restrict__ out, int NQ, int NK, int BHcnt)
{
    size_t id = (size_t)blockIdx.x * blockDim.x + threadIdx.x;
    size_t total = (size_t)BHcnt * NQ * HDIM;
    if (id >= total) return;
    int d = (int)(id % HDIM); size_t r = id / HDIM;
    int i = (int)(r % NQ);    r /= NQ;
    int h = (int)(r % HH);
    int bb = (int)(r / HH);
    const float* ar = attn + (((size_t)bb * HH + h) * NQ + i) * (size_t)NK;
    const float* vr = v + (size_t)bb * NK * ldv + h * HDIM + d;
    float s = 0.0f;
    for (int j = 0; j < NK; ++j) s += ar[j] * vr[(size_t)j * ldv];
    out[((size_t)bb * NQ + i) * DIMC + h * HDIM + d] = s;
}

// ---------------------------------------------------------------------------
// dep_merge kernels
// ---------------------------------------------------------------------------
__global__ __launch_bounds__(256)
void dist_kernel(const float* __restrict__ xcat, const float* __restrict__ xn,
                 float* __restrict__ dist, int Kp, int Nt, int C, int nb)
{
    size_t id = (size_t)blockIdx.x * blockDim.x + threadIdx.x;
    size_t total = (size_t)nb * Kp * Nt;
    if (id >= total) return;
    int t = (int)(id % Nt);
    int k = (int)((id / Nt) % Kp);
    int bb = (int)(id / ((size_t)Nt * Kp));
    const float* pr = xcat + ((size_t)bb * Nt + k) * C;   // protos = first Kp rows
    const float* xr = xcat + ((size_t)bb * Nt + t) * C;
    float dot = 0.0f;
    for (int c = 0; c < C; ++c) dot += pr[c] * xr[c];
    float d2 = xn[(size_t)bb * Nt + k] + xn[(size_t)bb * Nt + t] - 2.0f * dot;
    dist[id] = sqrtf(fmaxf(d2, 0.0f)) * rsqrtf((float)C);
}

__global__ __launch_bounds__(256)
void assign_kernel(const float* __restrict__ dist, int* __restrict__ idx,
                   float* __restrict__ csel, int Kp, int Nt, int nb)
{
    size_t id = (size_t)blockIdx.x * blockDim.x + threadIdx.x;
    size_t total = (size_t)nb * Nt;
    if (id >= total) return;
    int t = (int)(id % Nt);
    int bb = (int)(id / Nt);
    const float* dr = dist + (size_t)bb * Kp * Nt + t;
    const int P = Kp / 2;
    int best = 0; float bestd = dr[0];
    float c0 = 0.0f, c1 = 0.0f;
    for (int k = 0; k < Kp; ++k) {
        float d = dr[(size_t)k * Nt];
        if (d < bestd) { bestd = d; best = k; }
        float e = expf(-d);
        if (k < P) c0 += e; else c1 += e;
    }
    c0 /= (float)P; c1 /= (float)P;
    float denom = c0 + c1 + 1e-6f;
    idx[id] = best;
    csel[id] = (best < P) ? (c0 / denom) : (c1 / denom);
}

__global__ __launch_bounds__(256)
void accum_kernel(const int* __restrict__ idx, const float* __restrict__ csel,
                  const float* __restrict__ sw, float* __restrict__ allc,
                  float* __restrict__ alls, int Kp, int Nt, int nb)
{
    size_t id = (size_t)blockIdx.x * blockDim.x + threadIdx.x;
    size_t total = (size_t)nb * Nt;
    if (id >= total) return;
    int bb = (int)(id / Nt);
    int k = idx[id];
    atomicAdd(&allc[(size_t)bb * Kp + k], csel[id]);
    atomicAdd(&alls[(size_t)bb * Kp + k], sw[id]);
}

__global__ __launch_bounds__(256)
void weight_kernel(const int* __restrict__ idx, const float* __restrict__ csel,
                   const float* __restrict__ sw, const float* __restrict__ allc,
                   const float* __restrict__ alls, float* __restrict__ wtok,
                   int Kp, int Nt, int nb)
{
    size_t id = (size_t)blockIdx.x * blockDim.x + threadIdx.x;
    size_t total = (size_t)nb * Nt;
    if (id >= total) return;
    int bb = (int)(id / Nt);
    int k = idx[id];
    float sc = allc[(size_t)bb * Kp + k] + 1e-6f;
    float ss = alls[(size_t)bb * Kp + k] + 1e-6f;
    wtok[id] = 0.5f * csel[id] / sc + 0.5f * sw[id] / ss;
}

__global__ __launch_bounds__(256)
void scatter_merge_kernel(const float* __restrict__ xcat, const int* __restrict__ idx,
                          const float* __restrict__ w, float* __restrict__ merged,
                          int Kp, int Nt, int C, int nb)
{
    size_t id = (size_t)blockIdx.x * blockDim.x + threadIdx.x;
    size_t total = (size_t)nb * Nt * C;
    if (id >= total) return;
    int c = (int)(id % C); size_t r = id / C;
    int t = (int)(r % Nt);
    int bb = (int)(r / Nt);
    int k = idx[(size_t)bb * Nt + t];
    atomicAdd(&merged[((size_t)bb * Kp + k) * C + c],
              xcat[id] * w[(size_t)bb * Nt + t]);
}

// ---------------------------------------------------------------------------
// utility kernels
// ---------------------------------------------------------------------------
__global__ __launch_bounds__(256)
void copy_kernel(float* __restrict__ dst, const float* __restrict__ src, size_t n)
{
    size_t id = (size_t)blockIdx.x * blockDim.x + threadIdx.x;
    if (id < n) dst[id] = src[id];
}

__global__ __launch_bounds__(256)
void zero_kernel(float* __restrict__ p, size_t n)
{
    size_t id = (size_t)blockIdx.x * blockDim.x + threadIdx.x;
    if (id < n) p[id] = 0.0f;
}

__global__ __launch_bounds__(256)
void bcast_kernel(float* __restrict__ dst, const float* __restrict__ src, size_t per, int nb)
{
    size_t id = (size_t)blockIdx.x * blockDim.x + threadIdx.x;
    if (id < per * (size_t)nb) dst[id] = src[id % per];
}

__global__ __launch_bounds__(256)
void batchcopy_kernel(float* __restrict__ dst, const float* __restrict__ src,
                      size_t perBatch, size_t dstStride, size_t dstOff, int nb)
{
    size_t id = (size_t)blockIdx.x * blockDim.x + threadIdx.x;
    size_t total = (size_t)nb * perBatch;
    if (id >= total) return;
    size_t bb = id / perBatch;
    size_t r = id % perBatch;
    dst[bb * dstStride + dstOff + r] = src[id];
}

__global__ __launch_bounds__(256)
void maxpool_kernel(const float* __restrict__ p, float* __restrict__ out,
                    int Kp, int C, int nb)
{
    size_t id = (size_t)blockIdx.x * blockDim.x + threadIdx.x;
    int K2 = Kp / 2;
    size_t total = (size_t)nb * K2 * C;
    if (id >= total) return;
    int c = (int)(id % C); size_t r = id / C;
    int k2 = (int)(r % K2);
    int bb = (int)(r / K2);
    float a = p[((size_t)bb * Kp + 2 * k2) * C + c];
    float b = p[((size_t)bb * Kp + 2 * k2 + 1) * C + c];
    out[id] = fmaxf(a, b);
}

// ---------------------------------------------------------------------------
// host orchestration
// ---------------------------------------------------------------------------
static inline size_t cdivz(size_t a, size_t b) { return (a + b - 1) / b; }

static inline void gemm(const float* A, const float* W, const float* bias,
                        const float* res, float* C, int M, int K, int Nn,
                        int act, hipStream_t s)
{
    dim3 grid((unsigned)(Nn / GT_N), (unsigned)(M / GT_M));
    if (act == 1) {
        gemm_wmma_kernel<1, true, false><<<grid, 256, 0, s>>>(A, W, bias, nullptr, C, M, K, Nn);
    } else if (res) {
        gemm_wmma_kernel<0, true, true><<<grid, 256, 0, s>>>(A, W, bias, res, C, M, K, Nn);
    } else if (bias) {
        gemm_wmma_kernel<0, true, false><<<grid, 256, 0, s>>>(A, W, bias, nullptr, C, M, K, Nn);
    } else {
        gemm_wmma_kernel<0, false, false><<<grid, 256, 0, s>>>(A, W, nullptr, nullptr, C, M, K, Nn);
    }
}

extern "C" void kernel_launch(void* const* d_in, const int* in_sizes, int n_in,
                              void* d_out, int out_size, void* d_ws, size_t ws_size,
                              hipStream_t stream)
{
    (void)in_sizes; (void)n_in; (void)out_size; (void)ws_size;
    auto in = [&](int i) -> const float* { return (const float*)d_in[i]; };

    // flattened input indices (setup_inputs dict order, leaves depth-first)
    const int I_X = 0, I_PROTO = 1, I_FW = 2, I_FB = 6;
    const int I_T2I2 = 10, I_T2I4 = 28, I_L0 = 46, LSTRIDE = 16;
    enum { T_LN1G = 0, T_LN1B, T_LN2G, T_LN2B, T_FFG, T_FFB,
           T_QW, T_QB, T_KW, T_KB, T_VW, T_VB, T_PROJW, T_PROJB,
           T_FF1W, T_FF1B, T_FF2W, T_FF2B };
    enum { L_LN1G = 0, L_LN1B, L_LN2G, L_LN2B, L_FFG, L_FFB,
           L_TOQW, L_TOKVW, L_TOOUTW, L_TOOUTB,
           L_FF1W, L_FF1B, L_FF2W, L_FF2B, L_SCOREW, L_SCOREB };

    // workspace bump allocator
    char* base = (char*)d_ws;
    size_t off = 0;
    auto alloc = [&](size_t elems) -> float* {
        float* p = (float*)(base + off);
        off += elems * sizeof(float);
        off = (off + 255) & ~(size_t)255;
        return p;
    };

    const size_t XN = (size_t)BB * NTOK * DIMC;
    const size_t NtMax = 128 + NTOK;
    float* x0   = alloc(XN);
    float* x1   = alloc(XN);
    float* xln  = alloc(XN);
    float* qbuf = alloc(XN);                         // q / attn-out for token queries
    float* kbuf = alloc(XN);
    float* vbuf = alloc(XN);
    float* big  = alloc((size_t)BB * NTOK * MLPC);   // mlp hidden / attention dots
    float* xcat = alloc((size_t)BB * NtMax * DIMC);
    float* pA   = alloc((size_t)BB * 128 * DIMC);
    float* pB   = alloc((size_t)BB * 128 * DIMC);
    float* pLN  = alloc((size_t)BB * 128 * DIMC);
    float* pQ   = alloc((size_t)BB * 128 * DIMC);
    float* pO   = alloc((size_t)BB * 128 * DIMC);
    float* kv   = alloc((size_t)BB * 128 * 2 * DIMC);
    float* hsm  = alloc((size_t)BB * 128 * MLPC);
    float* sw   = alloc((size_t)BB * NtMax);
    float* xnrm = alloc((size_t)BB * NtMax);
    float* csel = alloc((size_t)BB * NtMax);
    float* wtok = alloc((size_t)BB * NtMax);
    int*   idxc = (int*)alloc((size_t)BB * NtMax);
    float* allc = alloc((size_t)BB * 128);
    float* alls = alloc((size_t)BB * 128);
    float* dst_ = alloc((size_t)BB * 128 * NtMax);

    // x ping-pong (never mutate d_in)
    copy_kernel<<<(unsigned)cdivz(XN, 256), 256, 0, stream>>>(x0, in(I_X), XN);
    float* xc = x0; float* xa = x1;

    // broadcast prototypes across batch
    int Kp = 8;
    bcast_kernel<<<(unsigned)cdivz((size_t)BB * Kp * DIMC, 256), 256, 0, stream>>>(
        pA, in(I_PROTO), (size_t)Kp * DIMC, BB);
    float* pc = pA; float* pa = pB;

    auto swap_p = [&]() { float* t = pc; pc = pa; pa = t; };
    auto swap_x = [&]() { float* t = xc; xc = xa; xa = t; };

    auto fission = [&](int fi) {
        gemm(pc, in(I_FW + fi), in(I_FB + fi), nullptr, pa, BB * Kp, DIMC, 2 * DIMC, 0, stream);
        swap_p(); Kp *= 2;
    };

    auto t2i_block = [&](int TB) {
        const int Mp = BB * Kp;
        ln_kernel<<<(unsigned)Mp, 256, 0, stream>>>(pc, in(TB + T_LN1G), in(TB + T_LN1B), pLN, DIMC);
        ln_kernel<<<(unsigned)(BB * NTOK), 256, 0, stream>>>(xc, in(TB + T_LN2G), in(TB + T_LN2B), xln, DIMC);
        gemm(pLN, in(TB + T_QW), in(TB + T_QB), nullptr, pQ, Mp, DIMC, DIMC, 0, stream);
        gemm(xln, in(TB + T_KW), in(TB + T_KB), nullptr, kbuf, BB * NTOK, DIMC, DIMC, 0, stream);
        gemm(xln, in(TB + T_VW), in(TB + T_VB), nullptr, vbuf, BB * NTOK, DIMC, DIMC, 0, stream);
        const size_t nd = (size_t)BB * HH * Kp * NTOK;
        dots_kernel<<<(unsigned)cdivz(nd, 256), 256, 0, stream>>>(
            pQ, DIMC, kbuf, DIMC, big, Kp, NTOK, 0.125f, BB * HH);
        softmax_kernel<<<(unsigned)(BB * HH * Kp), 256, 0, stream>>>(big, NTOK, 1);
        const size_t no = (size_t)BB * HH * Kp * HDIM;
        attnout_kernel<<<(unsigned)cdivz(no, 256), 256, 0, stream>>>(
            big, vbuf, DIMC, pO, Kp, NTOK, BB * HH);
        gemm(pO, in(TB + T_PROJW), in(TB + T_PROJB), pc, pa, Mp, DIMC, DIMC, 0, stream);
        swap_p();
        ln_kernel<<<(unsigned)Mp, 256, 0, stream>>>(pc, in(TB + T_FFG), in(TB + T_FFB), pLN, DIMC);
        gemm(pLN, in(TB + T_FF1W), in(TB + T_FF1B), nullptr, hsm, Mp, DIMC, MLPC, 1, stream);
        gemm(hsm, in(TB + T_FF2W), in(TB + T_FF2B), pc, pa, Mp, MLPC, DIMC, 0, stream);
        swap_p();
    };

    fission(0); fission(1);
    t2i_block(I_T2I2);
    fission(2); fission(3);
    t2i_block(I_T2I4);

    for (int li = 0; li < 6; ++li) {
        const int LB = I_L0 + li * LSTRIDE;
        const int Nt = Kp + NTOK;

        // ---- dep_merge ----
        batchcopy_kernel<<<(unsigned)cdivz((size_t)BB * Kp * DIMC, 256), 256, 0, stream>>>(
            xcat, pc, (size_t)Kp * DIMC, (size_t)Nt * DIMC, 0, BB);
        batchcopy_kernel<<<(unsigned)cdivz(XN, 256), 256, 0, stream>>>(
            xcat, xc, (size_t)NTOK * DIMC, (size_t)Nt * DIMC, (size_t)Kp * DIMC, BB);
        score_kernel<<<(unsigned)(BB * Nt), 256, 0, stream>>>(
            xcat, in(LB + L_SCOREW), in(LB + L_SCOREB), sw, DIMC);
        sqnorm_kernel<<<(unsigned)(BB * Nt), 256, 0, stream>>>(xcat, xnrm, DIMC);
        const size_t ndst = (size_t)BB * Kp * Nt;
        dist_kernel<<<(unsigned)cdivz(ndst, 256), 256, 0, stream>>>(
            xcat, xnrm, dst_, Kp, Nt, DIMC, BB);
        assign_kernel<<<(unsigned)cdivz((size_t)BB * Nt, 256), 256, 0, stream>>>(
            dst_, idxc, csel, Kp, Nt, BB);
        zero_kernel<<<(unsigned)cdivz((size_t)BB * Kp, 256), 256, 0, stream>>>(allc, (size_t)BB * Kp);
        zero_kernel<<<(unsigned)cdivz((size_t)BB * Kp, 256), 256, 0, stream>>>(alls, (size_t)BB * Kp);
        accum_kernel<<<(unsigned)cdivz((size_t)BB * Nt, 256), 256, 0, stream>>>(
            idxc, csel, sw, allc, alls, Kp, Nt, BB);
        weight_kernel<<<(unsigned)cdivz((size_t)BB * Nt, 256), 256, 0, stream>>>(
            idxc, csel, sw, allc, alls, wtok, Kp, Nt, BB);
        zero_kernel<<<(unsigned)cdivz((size_t)BB * Kp * DIMC, 256), 256, 0, stream>>>(
            pa, (size_t)BB * Kp * DIMC);
        scatter_merge_kernel<<<(unsigned)cdivz((size_t)BB * Nt * DIMC, 256), 256, 0, stream>>>(
            xcat, idxc, wtok, pa, Kp, Nt, DIMC, BB);
        swap_p();

        // ---- vit cross-attention: q from tokens, kv from prototypes ----
        ln_kernel<<<(unsigned)(BB * NTOK), 256, 0, stream>>>(xc, in(LB + L_LN1G), in(LB + L_LN1B), xln, DIMC);
        ln_kernel<<<(unsigned)(BB * Kp), 256, 0, stream>>>(pc, in(LB + L_LN2G), in(LB + L_LN2B), pLN, DIMC);
        gemm(xln, in(LB + L_TOQW), nullptr, nullptr, qbuf, BB * NTOK, DIMC, DIMC, 0, stream);
        gemm(pLN, in(LB + L_TOKVW), nullptr, nullptr, kv, BB * Kp, DIMC, 2 * DIMC, 0, stream);
        const size_t nd = (size_t)BB * HH * NTOK * Kp;
        dots_kernel<<<(unsigned)cdivz(nd, 256), 256, 0, stream>>>(
            qbuf, DIMC, kv, 2 * DIMC, big, NTOK, Kp, 0.125f, BB * HH);
        softmax_kernel<<<(unsigned)(BB * HH * NTOK), 256, 0, stream>>>(big, Kp, 0);
        const size_t no = (size_t)BB * HH * NTOK * HDIM;
        attnout_kernel<<<(unsigned)cdivz(no, 256), 256, 0, stream>>>(
            big, kv + DIMC, 2 * DIMC, qbuf, NTOK, Kp, BB * HH);
        gemm(qbuf, in(LB + L_TOOUTW), in(LB + L_TOOUTB), xc, xa, BB * NTOK, DIMC, DIMC, 0, stream);
        swap_x();

        // ---- feed-forward on tokens ----
        ln_kernel<<<(unsigned)(BB * NTOK), 256, 0, stream>>>(xc, in(LB + L_FFG), in(LB + L_FFB), xln, DIMC);
        gemm(xln, in(LB + L_FF1W), in(LB + L_FF1B), nullptr, big, BB * NTOK, DIMC, MLPC, 1, stream);
        gemm(big, in(LB + L_FF2W), in(LB + L_FF2B), xc, xa, BB * NTOK, MLPC, DIMC, 0, stream);
        swap_x();

        // ---- maxpool clusters after layer 3 ----
        if ((li + 1) % 3 == 0 && (li + 1) < 6) {
            maxpool_kernel<<<(unsigned)cdivz((size_t)BB * (Kp / 2) * DIMC, 256), 256, 0, stream>>>(
                pc, pa, Kp, DIMC, BB);
            swap_p(); Kp /= 2;
        }
    }

    copy_kernel<<<(unsigned)cdivz(XN, 256), 256, 0, stream>>>((float*)d_out, xc, XN);
}
